// HierarchicalPooling_76209899700419
// MI455X (gfx1250) — matrix operand order repaired and checked by reading, not compiled, not run on hardware
//
#include <hip/hip_runtime.h>
#include <hip/hip_bf16.h>

typedef __bf16 bf16_t;
typedef __attribute__((ext_vector_type(16))) __bf16 v16bf;
typedef __attribute__((ext_vector_type(8)))  float  v8f;

#define N_PTS   100000
#define C_DIM   128
#define K_CL    16
#define B_GR    16
#define TAUV    1.0f
#define EPSV    1e-9f
#define CHUNK   32
#define NCHUNKS (N_PTS / CHUNK)          /* 3125 exactly */
#define WAVES_PER_BLK 2
#define NBLOCKS 256
#define TOTAL_WAVES (NBLOCKS * WAVES_PER_BLK)   /* 512 */

/* ---- workspace layout ----
   bytes [0, 32768): W1 prepacked bf16 B-fragments (8 ntiles x 4 kchunks x 32 lanes x 16)
   bytes [32768, 36864): W2 prepacked bf16 B-fragments (4 kchunks x 32 lanes x 16)
   floats from WS_F_BASE: accumulators                                        */
#define WS_F_BASE  (36864 / 4)
#define OFF_COLSUM 0        /* 16  : sum_n s[n,k]                 */
#define OFF_SPOS   16       /* 48  : sum_n s[n,k]*pos[n,:]        */
#define OFF_SPOSSQ 64       /* 16  : sum_n s[n,k]*|pos[n]|^2      */
#define OFF_ENT    80       /* 1   : sum_n sum_k s log(s+eps)     */
#define OFF_SSP    81       /* 768 : per-batch sum s*pos [B,K,3]  */
#define OFF_SS     849      /* 256 : per-batch sum s     [B,K]    */
#define WS_F_COUNT 1105

/* ---- d_out layout (floats) ---- */
#define OUT_OFF  0            /* [B,K,C] = 32768      */
#define S_OFF    32768        /* [N,K]   = 1600000    */
#define LOSS_OFF 1632768      /* 5 scalars (alpha: diversity, entropy, pruning, separation, spatial) */
#define MU_OFF   1632773      /* [B,K,3] = 768        */

__global__ void hp_zero(float* out, float* wsf) {
    int i = blockIdx.x * blockDim.x + threadIdx.x;
    if (i < B_GR * K_CL * C_DIM) out[OUT_OFF + i] = 0.0f;
    if (i < WS_F_COUNT) wsf[i] = 0.0f;
}

/* Prepack W1 [128(k) x 128(n)] and W2 [128(k) x 16(n)] into bf16 WMMA
   B-fragment order: frag(t,q): lane L holds column n = 16t + (L&15);
   element e (0..15) holds K = 32q + 8*(L>=16) + (e&7) + 16*(e>=8).       */
__global__ void hp_prepack(const float* __restrict__ W1,
                           const float* __restrict__ W2,
                           bf16_t* __restrict__ wp) {
    int i = blockIdx.x * blockDim.x + threadIdx.x;
    if (i < 16384) {
        int e = i & 15, L = (i >> 4) & 31, q = (i >> 9) & 3, t = i >> 11;
        int n = t * 16 + (L & 15);
        int k = q * 32 + ((L >> 4) << 3) + (e & 7) + ((e >> 3) << 4);
        wp[i] = (bf16_t)W1[k * C_DIM + n];
    }
    if (i < 2048) {
        int e = i & 15, L = (i >> 4) & 31, q = i >> 9;
        int n = L & 15;
        int k = q * 32 + ((L >> 4) << 3) + (e & 7) + ((e >> 3) << 4);
        wp[16384 + i] = (bf16_t)W2[k * K_CL + n];
    }
}

__global__ __launch_bounds__(64) __attribute__((amdgpu_num_vgpr(256)))
void hp_main(
    const float* __restrict__ x, const float* __restrict__ pos,
    const int* __restrict__ batch, const float* __restrict__ gumbel,
    const float* __restrict__ b1, const float* __restrict__ b2,
    const float* __restrict__ scaling,
    const uint4* __restrict__ wsrc, float* __restrict__ wsf,
    float* __restrict__ out, float* __restrict__ s_out)
{
    __shared__ __align__(32) bf16_t lw[18432];                   /* 36 KB W1/W2 frags   */
    __shared__ __align__(16) float xs[WAVES_PER_BLK][2][4096];   /* 64 KB x dbl-buffer  */
    __shared__ __align__(16) float hs[WAVES_PER_BLK][2048];      /* 16 KB h stage       */

    for (int i = threadIdx.x; i < 2304; i += blockDim.x)
        ((uint4*)lw)[i] = wsrc[i];
    __syncthreads();

    const int lane = threadIdx.x & 31;
    const int wave = threadIdx.x >> 5;
    const int half = lane >> 4;
    const int lcol = lane & 15;
    const int h8   = half * 8;

    float* const xbuf0 = xs[wave][0];
    float* const xbuf1 = xs[wave][1];
    float* const hw    = hs[wave];
    const unsigned xbase0 = (unsigned)(size_t)xbuf0;   /* LDS byte offset (low 32 bits) */
    const unsigned xbase1 = (unsigned)(size_t)xbuf1;
    const float scal  = scaling[0];
    const float bias2 = b2[lcol];

    const int gw  = blockIdx.x * WAVES_PER_BLK + wave;
    const int per = (NCHUNKS + TOTAL_WAVES - 1) / TOTAL_WAVES;
    const int c0  = gw * per;
    const int c1  = (c0 + per < NCHUNKS) ? (c0 + per) : NCHUNKS;

    /* issue one 32x128 f32 tile as 32 async b128 DMA ops into LDS buffer p */
    auto issue_stage = [&](int chn, int p) {
        const float* xg = x + (size_t)chn * CHUNK * C_DIM;
        const unsigned base = p ? xbase1 : xbase0;
        #pragma unroll
        for (int i = 0; i < 32; ++i) {
            const unsigned off16 = (unsigned)((i * 32 + lane) * 16);
            asm volatile("global_load_async_to_lds_b128 %0, %1, %2"
                         :: "v"(base + off16), "v"(off16), "s"(xg) : "memory");
        }
    };

    v8f oacc[8];
    float segp0 = 0.f, segp1 = 0.f, segp2 = 0.f, segp3 = 0.f;
    float colp = 0.f, sp0 = 0.f, sp1 = 0.f, sp2 = 0.f, ssqp = 0.f, entp = 0.f;
    int  cur_b = -1;
    bool have  = false;

    /* flush per-batch accumulators (pooled tile + centroid sums) */
    auto flush_seg = [&](int bb) {
        #pragma unroll
        for (int t = 0; t < 8; ++t)
            #pragma unroll
            for (int j = 0; j < 8; ++j)
                atomicAdd(&out[OUT_OFF + ((bb * K_CL + j + h8) * C_DIM) + t * 16 + lcol], oacc[t][j]);
        atomicAdd(&wsf[OFF_SSP + (bb * K_CL + lcol) * 3 + 0], segp0);
        atomicAdd(&wsf[OFF_SSP + (bb * K_CL + lcol) * 3 + 1], segp1);
        atomicAdd(&wsf[OFF_SSP + (bb * K_CL + lcol) * 3 + 2], segp2);
        atomicAdd(&wsf[OFF_SS + bb * K_CL + lcol], segp3);
    };

    int p = 0;
    if (c0 < c1) issue_stage(c0, 0);

    for (int ch = c0; ch < c1; ++ch) {
        const int n0 = ch * CHUNK;

        /* double-buffered async pipeline: refill other buffer, then wait for ours.
           Async loads complete in order, so asynccnt<=32 => this chunk's 32 are done. */
        if (ch + 1 < c1) {
            issue_stage(ch + 1, p ^ 1);
            if (ch + 2 < c1) __builtin_prefetch(x + (size_t)(ch + 2) * CHUNK * C_DIM, 0, 1);
            asm volatile("s_wait_asynccnt 0x20" ::: "memory");
        } else {
            asm volatile("s_wait_asynccnt 0x0" ::: "memory");
        }
        float* const xsf = p ? xbuf1 : xbuf0;

        const int bfirst = batch[n0];
        const int blast  = batch[n0 + 31];

        float sval[16];
        float csp0 = 0.f, csp1 = 0.f, csp2 = 0.f, csp3 = 0.f;

        #pragma unroll
        for (int st = 0; st < 2; ++st) {
            /* x A-fragments for this 16-row subtile: built once, reused by 8 ntiles */
            v16bf a4[4];
            #pragma unroll
            for (int q = 0; q < 4; ++q) {
                const float* xr = xsf + (st * 16 + lcol) * C_DIM + q * 32 + h8;
                #pragma unroll
                for (int e = 0; e < 8; ++e) {
                    a4[q][e]     = (bf16_t)xr[e];
                    a4[q][e + 8] = (bf16_t)xr[16 + e];
                }
            }
            /* ---- GEMM1: h = relu(x@W1+b1) ---- */
            #pragma unroll
            for (int t = 0; t < 8; ++t) {
                const float bias = b1[t * 16 + lcol];
                v8f acc;
                #pragma unroll
                for (int j = 0; j < 8; ++j) acc[j] = bias;
                #pragma unroll
                for (int q = 0; q < 4; ++q) {
                    v16bf bfr = *(v16bf*)(lw + ((t * 4 + q) * 32 + lane) * 16);
                    acc = __builtin_amdgcn_wmma_f32_16x16x32_bf16(
                        false, a4[q], false, bfr, (short)0, acc, false, false);
                }
                #pragma unroll
                for (int j = 0; j < 8; ++j) {
                    float v = acc[j] > 0.f ? acc[j] : 0.f;
                    hw[(j + h8) * C_DIM + t * 16 + lcol] = v;
                }
            }
            asm volatile("s_wait_dscnt 0x0" ::: "memory");

            /* ---- GEMM2: logits = (h@W2+b2)*scaling ---- */
            v8f acc2;
            #pragma unroll
            for (int j = 0; j < 8; ++j) acc2[j] = bias2;
            #pragma unroll
            for (int q = 0; q < 4; ++q) {
                v16bf ha;
                const float* hr = hw + lcol * C_DIM + q * 32 + h8;
                #pragma unroll
                for (int e = 0; e < 8; ++e) {
                    ha[e]     = (bf16_t)hr[e];
                    ha[e + 8] = (bf16_t)hr[16 + e];
                }
                v16bf bfr = *(v16bf*)(lw + 16384 + (q * 32 + lane) * 16);
                acc2 = __builtin_amdgcn_wmma_f32_16x16x32_bf16(
                    false, ha, false, bfr, (short)0, acc2, false, false);
            }

            /* ---- gumbel-softmax (16-lane shuffle reduction) + stats ---- */
            #pragma unroll
            for (int j = 0; j < 8; ++j) {
                const int row = n0 + st * 16 + j + h8;
                float v = (acc2[j] * scal + gumbel[(size_t)row * K_CL + lcol]) * (1.0f / TAUV);
                float m = v;
                m = fmaxf(m, __shfl_xor(m, 1));
                m = fmaxf(m, __shfl_xor(m, 2));
                m = fmaxf(m, __shfl_xor(m, 4));
                m = fmaxf(m, __shfl_xor(m, 8));
                float ev = __expf(v - m);
                float sm = ev;
                sm += __shfl_xor(sm, 1);
                sm += __shfl_xor(sm, 2);
                sm += __shfl_xor(sm, 4);
                sm += __shfl_xor(sm, 8);
                float s = ev / sm;
                sval[st * 8 + j] = s;
                s_out[(size_t)row * K_CL + lcol] = s;
                entp += s * __logf(s + EPSV);
                colp += s;
                float px = pos[row * 3 + 0], py = pos[row * 3 + 1], pz = pos[row * 3 + 2];
                sp0 += s * px; sp1 += s * py; sp2 += s * pz;
                ssqp += s * (px * px + py * py + pz * pz);
                csp0 += s * px; csp1 += s * py; csp2 += s * pz; csp3 += s;
            }
        }

        if (bfirst == blast) {
            if (cur_b != bfirst) {
                if (have) flush_seg(cur_b);
                #pragma unroll
                for (int t = 0; t < 8; ++t)
                    #pragma unroll
                    for (int j = 0; j < 8; ++j) oacc[t][j] = 0.f;
                segp0 = segp1 = segp2 = segp3 = 0.f;
                cur_b = bfirst; have = true;
            }
            /* sval[] is already in sT A-fragment element order: no lane movement */
            v16bf as;
            #pragma unroll
            for (int e = 0; e < 16; ++e) as[e] = (bf16_t)sval[e];
            #pragma unroll
            for (int t = 0; t < 8; ++t) {
                v16bf xb;   /* x B-fragment built transiently per ntile */
                const float* xc = xsf + h8 * C_DIM + t * 16 + lcol;
                #pragma unroll
                for (int e = 0; e < 16; ++e) xb[e] = (bf16_t)xc[((e & 7) + ((e >> 3) << 4)) * C_DIM];
                oacc[t] = __builtin_amdgcn_wmma_f32_16x16x32_bf16(
                    false, as, false, xb, (short)0, oacc[t], false, false);
            }
            segp0 += csp0; segp1 += csp1; segp2 += csp2; segp3 += csp3;
        } else {
            /* segment boundary inside chunk (rare): flush then masked per-b */
            if (have) {
                flush_seg(cur_b);
                have = false; cur_b = -1;
                segp0 = segp1 = segp2 = segp3 = 0.f;
            }
            int bt[16];
            #pragma unroll
            for (int e = 0; e < 16; ++e)
                bt[e] = batch[n0 + (e >> 3) * 16 + (e & 7) + h8];
            for (int bb = bfirst; bb <= blast; ++bb) {
                v16bf as;
                float m0 = 0.f, m1 = 0.f, m2 = 0.f, m3 = 0.f;
                #pragma unroll
                for (int e = 0; e < 16; ++e) {
                    float sv = (bt[e] == bb) ? sval[e] : 0.f;
                    as[e] = (bf16_t)sv;
                    if (bt[e] == bb) {
                        int row = n0 + (e >> 3) * 16 + (e & 7) + h8;
                        float px = pos[row * 3 + 0], py = pos[row * 3 + 1], pz = pos[row * 3 + 2];
                        m0 += sv * px; m1 += sv * py; m2 += sv * pz; m3 += sv;
                    }
                }
                #pragma unroll
                for (int t = 0; t < 8; ++t) {
                    v16bf xb;
                    const float* xc = xsf + h8 * C_DIM + t * 16 + lcol;
                    #pragma unroll
                    for (int e = 0; e < 16; ++e) xb[e] = (bf16_t)xc[((e & 7) + ((e >> 3) << 4)) * C_DIM];
                    v8f z;
                    #pragma unroll
                    for (int j = 0; j < 8; ++j) z[j] = 0.f;
                    z = __builtin_amdgcn_wmma_f32_16x16x32_bf16(
                        false, as, false, xb, (short)0, z, false, false);
                    #pragma unroll
                    for (int j = 0; j < 8; ++j)
                        atomicAdd(&out[OUT_OFF + ((bb * K_CL + j + h8) * C_DIM) + t * 16 + lcol], z[j]);
                }
                atomicAdd(&wsf[OFF_SSP + (bb * K_CL + lcol) * 3 + 0], m0);
                atomicAdd(&wsf[OFF_SSP + (bb * K_CL + lcol) * 3 + 1], m1);
                atomicAdd(&wsf[OFF_SSP + (bb * K_CL + lcol) * 3 + 2], m2);
                atomicAdd(&wsf[OFF_SS + bb * K_CL + lcol], m3);
            }
        }
        p ^= 1;
    }

    if (have) flush_seg(cur_b);

    atomicAdd(&wsf[OFF_COLSUM + lcol], colp);
    atomicAdd(&wsf[OFF_SPOS + lcol * 3 + 0], sp0);
    atomicAdd(&wsf[OFF_SPOS + lcol * 3 + 1], sp1);
    atomicAdd(&wsf[OFF_SPOS + lcol * 3 + 2], sp2);
    atomicAdd(&wsf[OFF_SPOSSQ + lcol], ssqp);
    float er = entp;
    er += __shfl_xor(er, 16);
    er += __shfl_xor(er, 8);
    er += __shfl_xor(er, 4);
    er += __shfl_xor(er, 2);
    er += __shfl_xor(er, 1);
    if (lane == 0) atomicAdd(&wsf[OFF_ENT], er);
}

__global__ void hp_finalize(const float* __restrict__ wsf, float* __restrict__ out) {
    __shared__ float mu[256 * 3];
    __shared__ float red[256];
    __shared__ float s_spat[16], s_div[16], s_prun[16];
    const int tid = threadIdx.x;
    const int b = tid >> 4, k = tid & 15;

    float inv = 1.f / (wsf[OFF_SS + tid] + EPSV);
    float m0 = wsf[OFF_SSP + tid * 3 + 0] * inv;
    float m1 = wsf[OFF_SSP + tid * 3 + 1] * inv;
    float m2 = wsf[OFF_SSP + tid * 3 + 2] * inv;
    mu[tid * 3 + 0] = m0; mu[tid * 3 + 1] = m1; mu[tid * 3 + 2] = m2;
    out[MU_OFF + tid * 3 + 0] = m0;
    out[MU_OFF + tid * 3 + 1] = m1;
    out[MU_OFF + tid * 3 + 2] = m2;

    if (tid < 16) {
        float cs  = wsf[OFF_COLSUM + tid];
        float ivs = 1.f / (cs + EPSV);
        float p0 = wsf[OFF_SPOS + tid * 3 + 0] * ivs;
        float p1 = wsf[OFF_SPOS + tid * 3 + 1] * ivs;
        float p2 = wsf[OFF_SPOS + tid * 3 + 2] * ivs;
        s_spat[tid] = wsf[OFF_SPOSSQ + tid] * ivs - (p0 * p0 + p1 * p1 + p2 * p2);
        float avg = cs / (float)N_PTS;
        s_div[tid]  = avg * __logf(avg + EPSV);
        s_prun[tid] = fabsf(avg);
    }
    __syncthreads();

    float rep = 0.f;
    for (int k2 = 0; k2 < K_CL; ++k2) {
        if (k2 == k) continue;
        float dx = m0 - mu[(b * K_CL + k2) * 3 + 0];
        float dy = m1 - mu[(b * K_CL + k2) * 3 + 1];
        float dz = m2 - mu[(b * K_CL + k2) * 3 + 2];
        rep += 1.f / (dx * dx + dy * dy + dz * dz + 1.f);
    }
    red[tid] = rep;
    __syncthreads();
    for (int s = 128; s > 0; s >>= 1) {
        if (tid < s) red[tid] += red[tid + s];
        __syncthreads();
    }
    if (tid == 0) {
        float spat = 0.f, dv = 0.f, pr = 0.f;
        for (int i = 0; i < 16; ++i) { spat += s_spat[i]; dv += s_div[i]; pr += s_prun[i]; }
        out[LOSS_OFF + 0] = dv;                                 /* diversity  */
        out[LOSS_OFF + 1] = -wsf[OFF_ENT] / (float)N_PTS;       /* entropy    */
        out[LOSS_OFF + 2] = pr / (float)K_CL;                   /* pruning    */
        out[LOSS_OFF + 3] = red[0] / ((float)(K_CL * (K_CL - 1)) + EPSV); /* separation */
        out[LOSS_OFF + 4] = spat / (float)K_CL;                 /* spatial    */
    }
}

extern "C" void kernel_launch(void* const* d_in, const int* in_sizes, int n_in,
                              void* d_out, int out_size, void* d_ws, size_t ws_size,
                              hipStream_t stream) {
    const float* x       = (const float*)d_in[0];
    const float* pos     = (const float*)d_in[1];
    const int*   batch   = (const int*)d_in[2];
    const float* gumbel  = (const float*)d_in[3];
    const float* W1      = (const float*)d_in[4];
    const float* b1      = (const float*)d_in[5];
    const float* W2      = (const float*)d_in[6];
    const float* b2      = (const float*)d_in[7];
    const float* scaling = (const float*)d_in[8];

    float* out = (float*)d_out;
    float* wsf = (float*)d_ws + WS_F_BASE;

    hp_zero<<<(B_GR * K_CL * C_DIM + 255) / 256, 256, 0, stream>>>(out, wsf);
    hp_prepack<<<64, 256, 0, stream>>>(W1, W2, (bf16_t*)d_ws);
    hp_main<<<NBLOCKS, WAVES_PER_BLK * 32, 0, stream>>>(
        x, pos, batch, gumbel, b1, b2, scaling,
        (const uint4*)d_ws, wsf, out, out + S_OFF);
    hp_finalize<<<1, 256, 0, stream>>>(wsf, out);
}
